// TopKRouter_62878321214268
// MI455X (gfx1250) — matrix operand order repaired and checked by reading, not compiled, not run on hardware
//
#include <hip/hip_runtime.h>

// ---------- problem constants (B=4, S=4096, D=2048, E=64, K=8) ----------
#define TOKENS        16384         // B*S
#define DDIM          2048
#define NEXP          64
#define TOPK          8
#define WG_TOKENS     128           // tokens per workgroup (8 waves x 16)
#define NTHREADS      256
#define LOGIT_STRIDE  65            // padded to kill LDS bank conflicts

// output layout: topk_prob | topk_idx | new_expert_bias | new_ema_load
#define PROB_OFF  0
#define IDX_OFF   (TOKENS * TOPK)                 // 131072
#define BIAS_OFF  (2 * TOKENS * TOPK)             // 262144
#define EMA_OFF   (2 * TOKENS * TOPK + NEXP)      // 262208

// workspace layout: [64 float counts][pad to 256B][gate_w as f16, 256KB]
#define WS_CNT_OFF   0
#define WS_GATE_OFF  256

// LDS partition (dynamic shared memory, 320KB/WGP on CDNA5)
#define GATE_BYTES   (NEXP * DDIM * 2)                      // 262144 (f16 gate_w)
#define LOGIT_BYTES  (WG_TOKENS * LOGIT_STRIDE * 4)         // 33280
#define CNT_BYTES    (NEXP * 4)                             // 256
#define SMEM_BYTES   (GATE_BYTES + LOGIT_BYTES + CNT_BYTES) // 295680 <= 320KB

#define BTILE_STRIDE  (16 * DDIM / 2)                       // 16384 words per 16-expert tile

typedef __attribute__((ext_vector_type(16))) _Float16 v16h;
typedef __attribute__((ext_vector_type(8)))  float    v8f;
typedef __attribute__((ext_vector_type(8)))  unsigned v8u;

// fp32 pair -> packed f16x2 in one VALU op (v_cvt_pk_rtz_f16_f32)
__device__ __forceinline__ unsigned pk_f16(float lo, float hi) {
    return __builtin_bit_cast(unsigned, __builtin_amdgcn_cvt_pkrtz(lo, hi));
}

// compare-exchange on (value, index), larger-lexicographic to slot i
#define CE_PAIR(i, j)                                                      \
    {                                                                      \
        const bool sw = (m[j] > m[i]) || (m[j] == m[i] && mi[j] < mi[i]);  \
        const float tv = sw ? m[j] : m[i];                                 \
        const float uv = sw ? m[i] : m[j];                                 \
        const int   ti = sw ? mi[j] : mi[i];                               \
        const int   ui = sw ? mi[i] : mi[j];                               \
        m[i] = tv; m[j] = uv; mi[i] = ti; mi[j] = ui;                      \
    }

// ---------------------------------------------------------------------------
// Kernel 0: one-shot gate_w fp32 -> f16 (written to workspace, L2-resident)
// ---------------------------------------------------------------------------
extern "C" __global__ void gate_convert_kernel(const float* __restrict__ gw,
                                               unsigned* __restrict__ gf16)
{
    const int i = blockIdx.x * blockDim.x + threadIdx.x;   // packed-pair index
    if (i < (NEXP * DDIM) / 2) {
        const float2 g = ((const float2*)gw)[i];
        gf16[i] = pk_f16(g.x, g.y);
    }
}

// ---------------------------------------------------------------------------
// Kernel 1: fused gate GEMM (f16 WMMA, fp32 accum) + top-8 + softmax + counts
// ---------------------------------------------------------------------------
extern "C" __global__ void __launch_bounds__(NTHREADS)
router_wmma_kernel(const float* __restrict__ x,
                   const unsigned* __restrict__ gate_f16,
                   const float* __restrict__ expert_bias,
                   float* __restrict__ out,
                   float* __restrict__ cnt_global)
{
    extern __shared__ char smem[];
    unsigned* gsm  = (unsigned*)smem;                          // f16 gate_w [E][D], packed pairs
    float*    lsm  = (float*)(smem + GATE_BYTES);              // logits [128][65]
    unsigned* lcnt = (unsigned*)(smem + GATE_BYTES + LOGIT_BYTES);

    const int tid  = threadIdx.x;
    const int lane = tid & 31;
    const int wid  = tid >> 5;
    const int wgBase = blockIdx.x * WG_TOKENS;

    // ---- stage pre-converted f16 gate into LDS: pure b128 copy, no math ----
    {
        const uint4* g4 = (const uint4*)gate_f16;
        uint4*       s4 = (uint4*)gsm;
        #pragma unroll 4
        for (int i = tid; i < (NEXP * DDIM) / 8; i += NTHREADS) s4[i] = g4[i];
    }
    if (tid < NEXP) lcnt[tid] = 0u;
    __syncthreads();

    // ---- per-wave WMMA GEMM: 16 tokens x 64 experts, K = 2048 in steps of 32 ----
    const int mrow  = lane & 15;            // token row within wave tile / expert col within tile
    const int hiH   = lane >> 4;            // lane half
    const int abase = hiH ? 16 : 0;         // A-fragment K sub-offset
    const int khalf = hiH ? 16 : 0;         // B-fragment K half
    const int tok   = wgBase + wid * 16 + mrow;
    const float* xrow = x + (size_t)tok * DDIM;

    v8f c0 = {}, c1 = {}, c2 = {}, c3 = {};

    const unsigned* bbase = gsm + ((mrow * DDIM + khalf) >> 1);

    // B-tile double buffer: prologue loads K-step 0
    v8u cur[4], nxt[4];
    #pragma unroll
    for (int t = 0; t < 4; ++t) cur[t] = *(const v8u*)(bbase + t * BTILE_STRIDE);

    #pragma unroll 2
    for (int k0 = 0; k0 < DDIM; k0 += 32) {
        __builtin_prefetch(xrow + k0 + 512, 0, 1);   // global_prefetch_b8, speculative

        // A fragment: 16x32 f16 layout — lane half selects K sub-ranges
        const float4 f0 = *(const float4*)(xrow + k0 + abase);
        const float4 f1 = *(const float4*)(xrow + k0 + abase + 4);
        const float4 f2 = *(const float4*)(xrow + k0 + abase + 8);
        const float4 f3 = *(const float4*)(xrow + k0 + abase + 12);
        v8u au;
        au[0] = pk_f16(f0.x, f0.y); au[1] = pk_f16(f0.z, f0.w);
        au[2] = pk_f16(f1.x, f1.y); au[3] = pk_f16(f1.z, f1.w);
        au[4] = pk_f16(f2.x, f2.y); au[5] = pk_f16(f2.z, f2.w);
        au[6] = pk_f16(f3.x, f3.y); au[7] = pk_f16(f3.z, f3.w);
        const v16h a = __builtin_bit_cast(v16h, au);

        // issue NEXT iteration's B-tile loads before consuming current tiles
        const int kwn = ((k0 + 32) & (DDIM - 1)) >> 1;   // wraps on last step (in-bounds)
        #pragma unroll
        for (int t = 0; t < 4; ++t) nxt[t] = *(const v8u*)(bbase + kwn + t * BTILE_STRIDE);

        c0 = __builtin_amdgcn_wmma_f32_16x16x32_f16(false, a, false,
                 __builtin_bit_cast(v16h, cur[0]), (short)0, c0, false, false);
        c1 = __builtin_amdgcn_wmma_f32_16x16x32_f16(false, a, false,
                 __builtin_bit_cast(v16h, cur[1]), (short)0, c1, false, false);
        c2 = __builtin_amdgcn_wmma_f32_16x16x32_f16(false, a, false,
                 __builtin_bit_cast(v16h, cur[2]), (short)0, c2, false, false);
        c3 = __builtin_amdgcn_wmma_f32_16x16x32_f16(false, a, false,
                 __builtin_bit_cast(v16h, cur[3]), (short)0, c3, false, false);

        #pragma unroll
        for (int t = 0; t < 4; ++t) cur[t] = nxt[t];     // rotate (renamed by unroll)
    }

    // ---- spill logits(+bias) to padded LDS: C layout lanes0-15: M=r; lanes16-31: M=r+8 ----
    {
        const int madd = hiH ? 8 : 0;
        float* lrow = lsm + (wid * 16) * LOGIT_STRIDE;
        const v8f cc[4] = {c0, c1, c2, c3};
        #pragma unroll
        for (int t = 0; t < 4; ++t) {
            const float bn = expert_bias[t * 16 + mrow];
            #pragma unroll
            for (int r = 0; r < 8; ++r) {
                lrow[(r + madd) * LOGIT_STRIDE + t * 16 + mrow] = cc[t][r] + bn;
            }
        }
    }
    __syncthreads();

    // ---- top-8: thread PAIR per token; each thread scans 32 experts, then
    //      a bitonic merge of the two sorted 8-lists via wave32 shfl_xor ----
    {
        const int lt = tid >> 1;                 // token 0..127 within WG
        const int h  = tid & 1;                  // expert half: [32h, 32h+32)
        const int tokG = wgBase + lt;
        const float* lg = lsm + lt * LOGIT_STRIDE + 32 * h;

        float v[32];
        #pragma unroll
        for (int e = 0; e < 32; ++e) v[e] = lg[e];

        // local masked 8x argmax over 32 (strict '>' => lowest index on ties)
        unsigned msk = 0u;
        float sc[TOPK]; int ix[TOPK];
        #pragma unroll
        for (int p = 0; p < TOPK; ++p) {
            float best = -3.4e38f; int bi = 0;
            #pragma unroll
            for (int e = 0; e < 32; ++e) {
                const bool ok = (((msk >> e) & 1u) == 0u) && (v[e] > best);
                best = ok ? v[e] : best;
                bi   = ok ? e    : bi;
            }
            msk |= 1u << bi;
            sc[p] = best;
            ix[p] = 32 * h + bi;
        }

        // stage 1 of bitonic merge: m[j] = lex-max( A[j], B[7-j] ) across the pair
        float m[TOPK]; int mi[TOPK];
        #pragma unroll
        for (int j = 0; j < TOPK; ++j) {
            const float pv = __shfl_xor(sc[7 - j], 1, 32);
            const int   pi = __shfl_xor(ix[7 - j], 1, 32);
            const bool takeP = (pv > sc[j]) || (pv == sc[j] && pi < ix[j]);
            m[j]  = takeP ? pv : sc[j];
            mi[j] = takeP ? pi : ix[j];
        }
        // bitonic merge of 8 (m is bitonic, holds the union's top-8) -> descending
        CE_PAIR(0, 4) CE_PAIR(1, 5) CE_PAIR(2, 6) CE_PAIR(3, 7)
        CE_PAIR(0, 2) CE_PAIR(1, 3) CE_PAIR(4, 6) CE_PAIR(5, 7)
        CE_PAIR(0, 1) CE_PAIR(2, 3) CE_PAIR(4, 5) CE_PAIR(6, 7)

        if (h == 0) {
            const float mx = m[0];
            float s = 0.f, ev[TOPK];
            #pragma unroll
            for (int p = 0; p < TOPK; ++p) { ev[p] = __expf(m[p] - mx); s += ev[p]; }
            const float inv = 1.0f / s;

            #pragma unroll
            for (int p = 0; p < TOPK; ++p) {
                out[PROB_OFF + (size_t)tokG * TOPK + p] = ev[p] * inv;
                out[IDX_OFF  + (size_t)tokG * TOPK + p] = (float)mi[p];
                atomicAdd(&lcnt[mi[p]], 1u);
            }
        }
    }
    __syncthreads();

    if (tid < NEXP) atomicAdd(cnt_global + tid, (float)lcnt[tid]);
}

// ---------------------------------------------------------------------------
// Kernel 2: load / EMA / bias update (tiny)
// ---------------------------------------------------------------------------
extern "C" __global__ void router_finalize_kernel(const float* __restrict__ expert_bias,
                                                  const float* __restrict__ ema_load,
                                                  const float* __restrict__ cnt_global,
                                                  float* __restrict__ out)
{
    const int e = threadIdx.x;
    if (e < NEXP) {
        const float load  = cnt_global[e] * (1.0f / (float)TOKENS);
        const float nema  = ema_load[e] * 0.95f + load * 0.05f;
        const float nbias = expert_bias[e] + 0.01f * ((1.0f / (float)NEXP) - nema);
        out[BIAS_OFF + e] = nbias;
        out[EMA_OFF  + e] = nema;
    }
}

// ---------------------------------------------------------------------------
extern "C" void kernel_launch(void* const* d_in, const int* in_sizes, int n_in,
                              void* d_out, int out_size, void* d_ws, size_t ws_size,
                              hipStream_t stream)
{
    const float* x    = (const float*)d_in[0];   // [4,4096,2048] f32
    const float* gw   = (const float*)d_in[1];   // [64,2048] f32
    const float* bias = (const float*)d_in[2];   // [64] f32
    const float* ema  = (const float*)d_in[3];   // [64] f32
    float* out = (float*)d_out;

    float*    cnt  = (float*)((char*)d_ws + WS_CNT_OFF);     // 64 float expert counts
    unsigned* gf16 = (unsigned*)((char*)d_ws + WS_GATE_OFF); // f16 gate_w, 256KB

    (void)hipMemsetAsync(cnt, 0, NEXP * sizeof(float), stream);  // graph-capturable memset node

    hipLaunchKernelGGL(gate_convert_kernel,
                       dim3((NEXP * DDIM / 2 + 255) / 256), dim3(256), 0, stream,
                       gw, gf16);

    hipLaunchKernelGGL(router_wmma_kernel,
                       dim3(TOKENS / WG_TOKENS), dim3(NTHREADS), SMEM_BYTES, stream,
                       x, gf16, bias, out, cnt);

    hipLaunchKernelGGL(router_finalize_kernel,
                       dim3(1), dim3(64), 0, stream,
                       bias, ema, cnt, out);
}
